// TernaryLinear_76914274337119
// MI455X (gfx1250) — matrix operand order repaired and checked by reading, not compiled, not run on hardware
//
#include <hip/hip_runtime.h>

typedef _Float16 h16  __attribute__((ext_vector_type(16)));
typedef _Float16 h8   __attribute__((ext_vector_type(8)));
typedef _Float16 h4   __attribute__((ext_vector_type(4)));
typedef float    f16f __attribute__((ext_vector_type(16)));
typedef float    f8   __attribute__((ext_vector_type(8)));
typedef float    f4   __attribute__((ext_vector_type(4)));

#define M_DIM 8192
#define N_DIM 4096
#define K_DIM 4096
#define BM 128
#define BN 128
#define BK 64
#define LDH  72   // fp16 LDS row stride: 144 B (16B-aligned, bank-rotating)
#define LDSF 68   // fp32 LDS row stride: 272 B (fallback kernel)

// ---------- CDNA5 async global->LDS helpers (ASYNCcnt-tracked) ----------
__device__ __forceinline__ void async_b128(const void* gsrc, unsigned lds_byte) {
    asm volatile("global_load_async_to_lds_b128 %0, %1, off"
                 :: "v"(lds_byte), "v"((unsigned long long)(uintptr_t)gsrc)
                 : "memory");
}
__device__ __forceinline__ void wait_async0() {
    asm volatile("s_wait_asynccnt 0x0" ::: "memory");
}

// =======================================================================
// Pass 1: one-time fp32 -> fp16 conversion (x and W are exactly / nearly
// representable; W ternary and bias binary are exact in fp16).
// =======================================================================
__global__ __launch_bounds__(256) void cvt_f32_to_f16(
    const float* __restrict__ src, _Float16* __restrict__ dst, int n4)
{
    for (int i = blockIdx.x * blockDim.x + threadIdx.x; i < n4;
         i += gridDim.x * blockDim.x) {
        f4 v = ((const f4*)src)[i];
        ((h4*)dst)[i] = __builtin_convertvector(v, h4);
    }
}

// =======================================================================
// Pass 2: fp16 WMMA GEMM, async double-buffered LDS, zero inner-loop cvt.
// y = fmod(x @ W^T + bias, 2)
// =======================================================================
__global__ __launch_bounds__(256) void gemm_f16_wmma_async(
    const _Float16* __restrict__ x16,   // [M, K]
    const _Float16* __restrict__ w16,   // [N, K]
    const float*    __restrict__ bias,  // [N]
    float*          __restrict__ out)   // [M, N]
{
    __shared__ _Float16 As[2][BM * LDH];
    __shared__ _Float16 Bs[2][BN * LDH];

    const int tid     = threadIdx.x;
    const int lane    = tid & 31;     // wave32
    const int wave    = tid >> 5;     // 0..7
    const int waveM   = wave & 1;     // 2 (M) x 4 (N) wave grid; 64x32 per wave
    const int waveN   = wave >> 1;
    const int l15     = lane & 15;
    const int halfSel = lane >> 4;

    const int blockM = blockIdx.y * BM;
    const int blockN = blockIdx.x * BN;

    f8 acc[4][2];
    #pragma unroll
    for (int mi = 0; mi < 4; ++mi)
        #pragma unroll
        for (int ni = 0; ni < 2; ++ni)
            #pragma unroll
            for (int e = 0; e < 8; ++e)
                acc[mi][ni][e] = 0.0f;

    // Async-fill one K-stage: 8 halves (16B) per chunk, 8 chunks per row.
    auto stage = [&](int k0, int b) {
        #pragma unroll
        for (int it = 0; it < (BM * BK / 8) / 256; ++it) {   // 4 iters
            int idx = tid + it * 256;
            int row = idx >> 3;
            int c8  = (idx & 7) << 3;
            async_b128(x16 + (size_t)(blockM + row) * K_DIM + (k0 + c8),
                       (unsigned)(uintptr_t)&As[b][row * LDH + c8]);
        }
        #pragma unroll
        for (int it = 0; it < (BN * BK / 8) / 256; ++it) {   // 4 iters
            int idx = tid + it * 256;
            int row = idx >> 3;
            int c8  = (idx & 7) << 3;
            async_b128(w16 + (size_t)(blockN + row) * K_DIM + (k0 + c8),
                       (unsigned)(uintptr_t)&Bs[b][row * LDH + c8]);
        }
    };

    stage(0, 0);
    wait_async0();
    __syncthreads();

    for (int k0 = 0; k0 < K_DIM; k0 += BK) {
        const int cur = (k0 >> 6) & 1;
        const int nxt = cur ^ 1;
        if (k0 + BK < K_DIM) stage(k0 + BK, nxt);   // overlap with compute

        const _Float16* Ab = As[cur];
        const _Float16* Bb = Bs[cur];

        #pragma unroll
        for (int kk = 0; kk < BK; kk += 32) {
            // A fragments (16x32): lanes 0-15 K={kk..+7, kk+16..+23}; lanes 16-31 +8.
            h16 af[4];
            #pragma unroll
            for (int mi = 0; mi < 4; ++mi) {
                int row = waveM * 64 + mi * 16 + l15;
                const _Float16* p = &Ab[row * LDH + kk + halfSel * 8];
                h8 lo = *(const h8*)(p);
                h8 hi = *(const h8*)(p + 16);
                af[mi] = __builtin_shufflevector(lo, hi,
                    0, 1, 2, 3, 4, 5, 6, 7, 8, 9, 10, 11, 12, 13, 14, 15);
            }
            // B fragments (32x16): lanes 0-15 K={kk..+7, kk+16..+23}; lanes 16-31 +8.
            h16 bf[2];
            #pragma unroll
            for (int ni = 0; ni < 2; ++ni) {
                int col = waveN * 32 + ni * 16 + l15;
                const _Float16* p = &Bb[col * LDH + kk + halfSel * 8];
                h8 lo = *(const h8*)(p);
                h8 hi = *(const h8*)(p + 16);
                bf[ni] = __builtin_shufflevector(lo, hi,
                    0, 1, 2, 3, 4, 5, 6, 7, 8, 9, 10, 11, 12, 13, 14, 15);
            }
            #pragma unroll
            for (int mi = 0; mi < 4; ++mi)
                #pragma unroll
                for (int ni = 0; ni < 2; ++ni)
                    acc[mi][ni] = __builtin_amdgcn_wmma_f32_16x16x32_f16(
                        false, af[mi], false, bf[ni],
                        (short)0, acc[mi][ni], false, false);
        }

        wait_async0();
        __syncthreads();
    }

    #pragma unroll
    for (int ni = 0; ni < 2; ++ni) {
        int col = blockN + waveN * 32 + ni * 16 + l15;
        float bv = bias[col];
        #pragma unroll
        for (int mi = 0; mi < 4; ++mi) {
            #pragma unroll
            for (int r = 0; r < 8; ++r) {
                int row = blockM + waveM * 64 + mi * 16 + halfSel * 8 + r;
                float y = acc[mi][ni][r] + bv;
                out[(size_t)row * N_DIM + col] = y - 2.0f * truncf(y * 0.5f);
            }
        }
    }
}

// =======================================================================
// Fallback (ws too small): fused fp32-LDS async kernel (round-2 version).
// =======================================================================
__device__ __forceinline__ h16 frag_cvt(f4 c0, f4 c1, f4 c2, f4 c3) {
    f8 lo = __builtin_shufflevector(c0, c1, 0, 1, 2, 3, 4, 5, 6, 7);
    f8 hi = __builtin_shufflevector(c2, c3, 0, 1, 2, 3, 4, 5, 6, 7);
    f16f full = __builtin_shufflevector(lo, hi,
        0, 1, 2, 3, 4, 5, 6, 7, 8, 9, 10, 11, 12, 13, 14, 15);
    return __builtin_convertvector(full, h16);
}

__global__ __launch_bounds__(128) void ternary_linear_wmma_fused(
    const float* __restrict__ x, const float* __restrict__ w,
    const float* __restrict__ bias, float* __restrict__ out)
{
    __shared__ float Abuf[2][BM * LDSF];
    __shared__ float Bbuf[2][BN * LDSF];

    const int tid = threadIdx.x, lane = tid & 31, wave = tid >> 5;
    const int waveM = wave & 1, waveN = wave >> 1;
    const int l15 = lane & 15, halfSel = lane >> 4;
    const int blockM = blockIdx.y * BM, blockN = blockIdx.x * BN;

    f8 acc[4][4];
    #pragma unroll
    for (int mi = 0; mi < 4; ++mi)
        #pragma unroll
        for (int ni = 0; ni < 4; ++ni)
            #pragma unroll
            for (int e = 0; e < 8; ++e) acc[mi][ni][e] = 0.0f;

    auto stage = [&](int k0, int b) {
        #pragma unroll
        for (int it = 0; it < (BM * BK / 4) / 128; ++it) {
            int idx = tid + it * 128, row = idx >> 4, c4 = (idx & 15) << 2;
            async_b128(x + (size_t)(blockM + row) * K_DIM + (k0 + c4),
                       (unsigned)(uintptr_t)&Abuf[b][row * LDSF + c4]);
        }
        #pragma unroll
        for (int it = 0; it < (BN * BK / 4) / 128; ++it) {
            int idx = tid + it * 128, row = idx >> 4, c4 = (idx & 15) << 2;
            async_b128(w + (size_t)(blockN + row) * K_DIM + (k0 + c4),
                       (unsigned)(uintptr_t)&Bbuf[b][row * LDSF + c4]);
        }
    };

    stage(0, 0);
    wait_async0();
    __syncthreads();

    for (int k0 = 0; k0 < K_DIM; k0 += BK) {
        const int cur = (k0 >> 6) & 1, nxt = cur ^ 1;
        if (k0 + BK < K_DIM) stage(k0 + BK, nxt);
        const float* Ab = Abuf[cur];
        const float* Bb = Bbuf[cur];

        #pragma unroll
        for (int kk = 0; kk < BK; kk += 32) {
            h16 af[4], bf[4];
            #pragma unroll
            for (int mi = 0; mi < 4; ++mi) {
                int row = waveM * 64 + mi * 16 + l15;
                const float* p = &Ab[row * LDSF + kk + halfSel * 8];
                af[mi] = frag_cvt(*(const f4*)(p),      *(const f4*)(p + 4),
                                  *(const f4*)(p + 16), *(const f4*)(p + 20));
            }
            #pragma unroll
            for (int ni = 0; ni < 4; ++ni) {
                int col = waveN * 64 + ni * 16 + l15;
                const float* p = &Bb[col * LDSF + kk + halfSel * 8];
                bf[ni] = frag_cvt(*(const f4*)(p),      *(const f4*)(p + 4),
                                  *(const f4*)(p + 16), *(const f4*)(p + 20));
            }
            #pragma unroll
            for (int mi = 0; mi < 4; ++mi)
                #pragma unroll
                for (int ni = 0; ni < 4; ++ni)
                    acc[mi][ni] = __builtin_amdgcn_wmma_f32_16x16x32_f16(
                        false, af[mi], false, bf[ni],
                        (short)0, acc[mi][ni], false, false);
        }
        wait_async0();
        __syncthreads();
    }

    #pragma unroll
    for (int ni = 0; ni < 4; ++ni) {
        int col = blockN + waveN * 64 + ni * 16 + l15;
        float bv = bias[col];
        #pragma unroll
        for (int mi = 0; mi < 4; ++mi)
            #pragma unroll
            for (int r = 0; r < 8; ++r) {
                int row = blockM + waveM * 64 + mi * 16 + halfSel * 8 + r;
                float y = acc[mi][ni][r] + bv;
                out[(size_t)row * N_DIM + col] = y - 2.0f * truncf(y * 0.5f);
            }
    }
}

// =======================================================================
extern "C" void kernel_launch(void* const* d_in, const int* in_sizes, int n_in,
                              void* d_out, int out_size, void* d_ws, size_t ws_size,
                              hipStream_t stream) {
    const float* x    = (const float*)d_in[0];   // [8192, 4096]
    const float* w    = (const float*)d_in[1];   // [4096, 4096]
    const float* bias = (const float*)d_in[2];   // [4096]
    float* out        = (float*)d_out;           // [8192, 4096]

    const size_t X16_BYTES = (size_t)M_DIM * K_DIM * sizeof(_Float16); // 64 MB
    const size_t W16_BYTES = (size_t)N_DIM * K_DIM * sizeof(_Float16); // 32 MB

    dim3 grid(N_DIM / BN, M_DIM / BM);           // (32, 64)

    if (ws_size >= X16_BYTES + W16_BYTES) {
        _Float16* x16 = (_Float16*)d_ws;
        _Float16* w16 = (_Float16*)((char*)d_ws + X16_BYTES);
        cvt_f32_to_f16<<<4096, 256, 0, stream>>>(x, x16, M_DIM * K_DIM / 4);
        cvt_f32_to_f16<<<2048, 256, 0, stream>>>(w, w16, N_DIM * K_DIM / 4);
        gemm_f16_wmma_async<<<grid, 256, 0, stream>>>(x16, w16, bias, out);
    } else {
        ternary_linear_wmma_fused<<<grid, 128, 0, stream>>>(x, w, bias, out);
    }
    (void)in_sizes; (void)n_in; (void)out_size;
}
